// DRNA_Block_43980465111466
// MI455X (gfx1250) — matrix operand order repaired and checked by reading, not compiled, not run on hardware
//
#include <hip/hip_runtime.h>

typedef __attribute__((ext_vector_type(16))) _Float16 v16h;
typedef __attribute__((ext_vector_type(8)))  _Float16 v8h;
typedef __attribute__((ext_vector_type(4)))  _Float16 h4;
typedef __attribute__((ext_vector_type(8)))  float    v8f;

#define DEV __device__ __forceinline__

constexpr int Bsz   = 2;
constexpr int Seq   = 2048;
constexpr int Dm    = 2048;
constexpr int Nh    = 16;
constexpr int Dh    = 128;
constexpr int Dff   = 8192;
constexpr int Mrows = Bsz * Seq;     // 4096 tokens
constexpr int QKVld = 3 * Dm;        // 6144
constexpr int PadK  = 40;            // padded K-stride for transposed LDS tiles

// ---------------------------------------------------------------------------
// WMMA helpers (CDNA5 gfx1250, wave32).  Layouts per CDNA5 ISA 7.12.2.
// ---------------------------------------------------------------------------
DEV v8f wmma16(v16h a, v16h b, v8f c) {
  return __builtin_amdgcn_wmma_f32_16x16x32_f16(false, a, false, b,
                                                (short)0, c, false, false);
}

DEV v8h ld8(const _Float16* p) { return *(const v8h*)p; }

DEV v16h concat16(v8h lo, v8h hi) {
  return __builtin_shufflevector(lo, hi, 0, 1, 2, 3, 4, 5, 6, 7,
                                         8, 9, 10, 11, 12, 13, 14, 15);
}

// A fragment: 16x32 f16 tile from row-major memory (leading dim ld, halves).
DEV v16h load_a_frag(const _Float16* src, int ld, int lane) {
  const int m    = lane & 15;
  const int kOff = (lane >> 4) << 3;          // 0 | 8
  const _Float16* p = src + m * ld + kOff;
  return concat16(ld8(p), ld8(p + 16));
}

// B fragment: 32x16 (K x N) where logical B[k][n] = base[n*ldRow + k].
DEV v16h load_bfragT(const _Float16* base, int ldRow, int lane) {
  const int n    = lane & 15;
  const int kOff = (lane >> 4) << 3;          // 0 | 8
  const _Float16* p = base + n * ldRow + kOff;
  return concat16(ld8(p), ld8(p + 16));
}

// Async copy of 32 contiguous bytes global->LDS (2x b128, ASYNCcnt-tracked).
// INST_OFFSET is added to BOTH the LDS and global addresses (ISA 15.18.3).
DEV void async_stage32B(unsigned ldsAddr, const _Float16* g) {
  asm volatile(
      "global_load_async_to_lds_b128 %0, %1, off\n\t"
      "global_load_async_to_lds_b128 %0, %1, off offset:16"
      :: "v"(ldsAddr), "v"((unsigned long long)(uintptr_t)g) : "memory");
}

DEV void wait_async0() { asm volatile("s_wait_asynccnt 0" ::: "memory"); }

// ---------------------------------------------------------------------------
// fp32 -> fp16 conversion (weights, once per launch)
// ---------------------------------------------------------------------------
__global__ __launch_bounds__(256) void cvt_f32_f16_kernel(
    const float* __restrict__ in, _Float16* __restrict__ out, int n) {
  for (int i = blockIdx.x * 256 + threadIdx.x; i < n; i += gridDim.x * 256)
    out[i] = (_Float16)in[i];
}

// ---------------------------------------------------------------------------
// LayerNorm: one row (D=2048) per 256-thread block, f16 output
// ---------------------------------------------------------------------------
__global__ __launch_bounds__(256) void layernorm_f16_kernel(
    const float* __restrict__ x, const float* __restrict__ g,
    const float* __restrict__ beta, _Float16* __restrict__ out, int D) {
  const int row = blockIdx.x;
  const int tid = threadIdx.x;
  const float* xr = x + (size_t)row * D;
  float s = 0.f, ss = 0.f;
  for (int i = tid; i < D; i += 256) {
    const float v = xr[i];
    s += v; ss += v * v;
  }
  __shared__ float rs[256], rq[256];
  rs[tid] = s; rq[tid] = ss;
  __syncthreads();
  for (int st = 128; st > 0; st >>= 1) {
    if (tid < st) { rs[tid] += rs[tid + st]; rq[tid] += rq[tid + st]; }
    __syncthreads();
  }
  const float mean = rs[0] / D;
  const float var  = rq[0] / D - mean * mean;
  const float rstd = rsqrtf(var + 1e-5f);
  for (int i = tid; i < D; i += 256)
    out[(size_t)row * D + i] = (_Float16)((xr[i] - mean) * rstd * g[i] + beta[i]);
}

// ---------------------------------------------------------------------------
// RoPE, in place on f16 q and k halves of the packed qkv activation
// ---------------------------------------------------------------------------
__global__ __launch_bounds__(256) void rope_kernel(
    _Float16* __restrict__ qkv, const float* __restrict__ cosT,
    const float* __restrict__ sinT) {
  const int idx   = blockIdx.x * 256 + threadIdx.x;
  const int total = Mrows * Nh * (Dh / 2);
  if (idx >= total) return;
  const int d = idx & 63;
  const int h = (idx >> 6) & (Nh - 1);
  const int t = idx >> 10;
  const int s = t & (Seq - 1);
  const float c   = cosT[s * Dh + d];
  const float sn  = sinT[s * Dh + d];
  const float c2  = cosT[s * Dh + d + 64];
  const float sn2 = sinT[s * Dh + d + 64];
  const size_t base = (size_t)t * QKVld + h * Dh;
#pragma unroll
  for (int which = 0; which < 2; ++which) {  // q then k
    _Float16* p = qkv + base + which * Dm;
    const float x1 = (float)p[d];
    const float x2 = (float)p[d + 64];
    p[d]      = (_Float16)(x1 * c  - x2 * sn);
    p[d + 64] = (_Float16)(x2 * c2 + x1 * sn2);
  }
}

// ---------------------------------------------------------------------------
// WMMA GEMM: out = epilogue(A[MxK] @ W[KxN] + bias)
//   Block tile 64x128, 4 waves (2x2), wave tile 32x64 -> 8 WMMAs / K-step
//   sharing two A fragments (1.5 LDS loads per WMMA).
//   A tile staged via GLOBAL_LOAD_ASYNC_TO_LDS_B128; W tile transposed in LDS.
// ---------------------------------------------------------------------------
__global__ __launch_bounds__(128) void gemm_f16_kernel(
    const _Float16* __restrict__ A, const _Float16* __restrict__ Bw,
    const float* __restrict__ bias, const float* __restrict__ residual,
    float* __restrict__ outF32, _Float16* __restrict__ outF16,
    int M, int N, int K, int applyGelu) {
  (void)M;
  __shared__ __align__(16) _Float16 sA[64][32];       //  4 KB row-major
  __shared__ __align__(16) _Float16 sBt[128][PadK];   // 10 KB [n][k] transposed

  const int nTiles = N >> 7;
  const int mBlk = blockIdx.x / nTiles;
  const int nBlk = blockIdx.x % nTiles;
  const int tid  = threadIdx.x;
  const int lane = tid & 31;
  const int wave = tid >> 5;
  const int wm = (wave >> 1) << 5;   // 0 | 32
  const int wn = (wave & 1) << 6;    // 0 | 64

  const v8f vzero = {};
  v8f acc[2][4];
#pragma unroll
  for (int i = 0; i < 2; ++i)
#pragma unroll
    for (int j = 0; j < 4; ++j) acc[i][j] = vzero;

  // A staging: 128 threads x 32 contiguous bytes, async to LDS
  const int ar = tid >> 1, ac = (tid & 1) << 4;
  const unsigned ldsA = (unsigned)(uintptr_t)&sA[ar][ac];
  const _Float16* gAk = A + (size_t)(mBlk * 64 + ar) * K + ac;
  // B staging: 2x (4x4 register transpose); cols bc..bc+3, rows br(+16)..+3
  const int bc = (tid & 31) << 2;    // 32 col-groups * 4 = 128
  const int br = (tid >> 5) << 2;    //  4 row-groups * 4 = 16 (x2 via s)
  const _Float16* gBk = Bw + (size_t)br * N + nBlk * 128 + bc;
  const size_t strideB = (size_t)32 * N;

  for (int k0 = 0; k0 < K; k0 += 32) {
    async_stage32B(ldsA, gAk);
#pragma unroll
    for (int s = 0; s < 2; ++s) {
      h4 rr[4];
#pragma unroll
      for (int i = 0; i < 4; ++i)
        rr[i] = *(const h4*)(gBk + (size_t)(16 * s + i) * N);
#pragma unroll
      for (int j = 0; j < 4; ++j) {
        h4 cv = { rr[0][j], rr[1][j], rr[2][j], rr[3][j] };
        *(h4*)&sBt[bc + j][br + 16 * s] = cv;
      }
    }
    if (k0 + 32 < K) {                    // global_prefetch_b8 next K-tile
      __builtin_prefetch(gAk + 32, 0, 0);
      __builtin_prefetch(gBk + strideB, 0, 0);
    }
    wait_async0();
    __syncthreads();

    const v16h a0 = load_a_frag(&sA[wm][0],      32, lane);
    const v16h a1 = load_a_frag(&sA[wm + 16][0], 32, lane);
#pragma unroll
    for (int j = 0; j < 4; ++j) {
      const v16h b = load_bfragT(&sBt[wn + j * 16][0], PadK, lane);
      acc[0][j] = wmma16(a0, b, acc[0][j]);
      acc[1][j] = wmma16(a1, b, acc[1][j]);
    }
    __syncthreads();
    gAk += 32;
    gBk += strideB;
  }

  const int rowHalf = (lane >> 4) << 3;
  const int nLow    = lane & 15;
#pragma unroll
  for (int i = 0; i < 2; ++i) {
#pragma unroll
    for (int j = 0; j < 4; ++j) {
      const int col = nBlk * 128 + wn + j * 16 + nLow;
      const float bv = bias ? bias[col] : 0.0f;
#pragma unroll
      for (int r = 0; r < 8; ++r) {
        const int row = mBlk * 64 + wm + i * 16 + r + rowHalf;
        float v = acc[i][j][r] + bv;
        if (applyGelu) v = 0.5f * v * (1.0f + erff(v * 0.70710678118654752f));
        if (residual)  v += residual[(size_t)row * N + col];
        if (outF32) outF32[(size_t)row * N + col] = v;
        if (outF16) outF16[(size_t)row * N + col] = (_Float16)v;
      }
    }
  }
}

// ---------------------------------------------------------------------------
// Flash attention, causal.  Block = 8 waves x 16 queries.  32 keys / iter.
// K tile async-staged row-major; V tile transposed in LDS.
// ---------------------------------------------------------------------------
__global__ __launch_bounds__(256) void flash_attn_kernel(
    const _Float16* __restrict__ qkv, _Float16* __restrict__ aout) {
  __shared__ __align__(16) _Float16 sK[32][Dh];       //  8 KB  [key][dh]
  __shared__ __align__(16) _Float16 sVt[Dh][PadK];    // 10 KB  [dh][key]
  __shared__ __align__(16) _Float16 sP[8][16][32];    // 8 KB  per-wave scratch

  const int qBlk = blockIdx.x;
  const int h    = blockIdx.y;
  const int b    = blockIdx.z;
  const int tid  = threadIdx.x;
  const int lane = tid & 31;
  const int wave = tid >> 5;

  const int q0 = qBlk * 128;
  const int qw = q0 + wave * 16;
  const int rowHalf = (lane >> 4) << 3;
  const int nLow    = lane & 15;
  const float scale = 0.08838834764831845f;   // 1/sqrt(Dh)

  // Q fragments, loaded once from global in A-fragment pattern
  const _Float16* qbase = qkv + (size_t)(b * Seq + qw) * QKVld + h * Dh;
  v16h aQ[4];
#pragma unroll
  for (int c = 0; c < 4; ++c) aQ[c] = load_a_frag(qbase + c * 32, QKVld, lane);

  const v8f vzero = {};
  v8f o[8];
#pragma unroll
  for (int j = 0; j < 8; ++j) o[j] = vzero;
  float mrow[8], lrow[8];
#pragma unroll
  for (int r = 0; r < 8; ++r) { mrow[r] = -1e30f; lrow[r] = 0.f; }

  // K staging: 256 threads x 32 contiguous bytes (async)
  const int stR = tid >> 3, stC = (tid & 7) << 4;
  const unsigned ldsK = (unsigned)(uintptr_t)&sK[stR][stC];
  const _Float16* kgk =
      qkv + (size_t)(b * Seq + stR) * QKVld + Dm + h * Dh + stC;
  // V staging: 4x4 register transpose into sVt[dh][key]
  const int vd0 = (tid & 31) << 2;
  const int vk0 = (tid >> 5) << 2;
  const _Float16* vgk =
      qkv + (size_t)(b * Seq + vk0) * QKVld + 2 * Dm + h * Dh + vd0;
  const size_t strideKV = (size_t)32 * QKVld;

  const int nkt = (q0 >> 5) + 4;     // uniform trip count for the block
  for (int kt = 0; kt < nkt; ++kt) {
    async_stage32B(ldsK, kgk);
    {
      h4 rr[4];
#pragma unroll
      for (int i = 0; i < 4; ++i)
        rr[i] = *(const h4*)(vgk + (size_t)i * QKVld);
#pragma unroll
      for (int j = 0; j < 4; ++j) {
        h4 cv = { rr[0][j], rr[1][j], rr[2][j], rr[3][j] };
        *(h4*)&sVt[vd0 + j][vk0] = cv;
      }
    }
    wait_async0();
    __syncthreads();

    // scores S = Q K^T : two 16x16 C tiles over 32 keys, dh in 4 chunks
    v8f s0 = vzero, s1 = vzero;
#pragma unroll
    for (int c = 0; c < 4; ++c) {
      s0 = wmma16(aQ[c], load_bfragT(&sK[0][c * 32],  Dh, lane), s0);
      s1 = wmma16(aQ[c], load_bfragT(&sK[16][c * 32], Dh, lane), s1);
    }

    // online softmax; row r of the C tiles = query qw + r + rowHalf
    const int key0 = kt * 32 + nLow;
#pragma unroll
    for (int r = 0; r < 8; ++r) {
      const int q = qw + r + rowHalf;
      float v0 = (float)s0[r] * scale;
      float v1 = (float)s1[r] * scale;
      if (key0 > q)      v0 = -1e30f;       // causal mask
      if (key0 + 16 > q) v1 = -1e30f;
      float mx = fmaxf(v0, v1);
#pragma unroll
      for (int off = 1; off < 16; off <<= 1)
        mx = fmaxf(mx, __shfl_xor(mx, off, 32));
      const float mnew  = fmaxf(mrow[r], mx);
      const float alpha = __expf(mrow[r] - mnew);
      const float p0 = __expf(v0 - mnew);
      const float p1 = __expf(v1 - mnew);
      float ps = p0 + p1;
#pragma unroll
      for (int off = 1; off < 16; off <<= 1)
        ps += __shfl_xor(ps, off, 32);
      lrow[r] = lrow[r] * alpha + ps;
      mrow[r] = mnew;
#pragma unroll
      for (int j = 0; j < 8; ++j) o[j][r] *= alpha;
      sP[wave][r + rowHalf][nLow]      = (_Float16)p0;
      sP[wave][r + rowHalf][nLow + 16] = (_Float16)p1;
    }
    // wave-local DS RAW hazard: stores above -> loads below (same wave)
    asm volatile("s_wait_dscnt 0" ::: "memory");

    const v16h aP = load_a_frag(&sP[wave][0][0], 32, lane);
#pragma unroll
    for (int j = 0; j < 8; ++j)
      o[j] = wmma16(aP, load_bfragT(&sVt[j * 16][0], PadK, lane), o[j]);

    __syncthreads();   // protect sK/sVt before next staging
    kgk += strideKV;
    vgk += strideKV;
  }

  float inv[8];
#pragma unroll
  for (int r = 0; r < 8; ++r) inv[r] = 1.0f / lrow[r];
#pragma unroll
  for (int j = 0; j < 8; ++j) {
#pragma unroll
    for (int r = 0; r < 8; ++r) {
      const int row = qw + r + rowHalf;
      const int col = h * Dh + j * 16 + nLow;
      aout[(size_t)(b * Seq + row) * Dm + col] = (_Float16)(o[j][r] * inv[r]);
    }
  }
}

// ---------------------------------------------------------------------------
// Host-side orchestration
// ---------------------------------------------------------------------------
extern "C" void kernel_launch(void* const* d_in, const int* in_sizes, int n_in,
                              void* d_out, int out_size, void* d_ws, size_t ws_size,
                              hipStream_t stream) {
  (void)in_sizes; (void)n_in; (void)out_size; (void)ws_size;
  const float* x     = (const float*)d_in[0];
  const float* cosT  = (const float*)d_in[1];
  const float* sinT  = (const float*)d_in[2];
  /* d_in[3] = additive mask: causal mask applied analytically, unused */
  const float* ln1_g = (const float*)d_in[4];
  const float* ln1_b = (const float*)d_in[5];
  const float* w_qkv = (const float*)d_in[6];
  const float* b_qkv = (const float*)d_in[7];
  const float* w_out = (const float*)d_in[8];
  const float* b_out = (const float*)d_in[9];
  const float* ln2_g = (const float*)d_in[10];
  const float* ln2_b = (const float*)d_in[11];
  const float* w_ff1 = (const float*)d_in[12];
  const float* b_ff1 = (const float*)d_in[13];
  const float* w_ff2 = (const float*)d_in[14];
  const float* b_ff2 = (const float*)d_in[15];

  char* ws = (char*)d_ws;
  size_t off = 0;
  auto alloc = [&](size_t bytes) -> void* {
    void* p = ws + off;
    off += (bytes + 255) & ~(size_t)255;
    return p;
  };
  _Float16* w_qkv_h = (_Float16*)alloc((size_t)Dm * 3 * Dm * 2);
  _Float16* w_out_h = (_Float16*)alloc((size_t)Dm * Dm * 2);
  _Float16* w_ff1_h = (_Float16*)alloc((size_t)Dm * Dff * 2);
  _Float16* w_ff2_h = (_Float16*)alloc((size_t)Dff * Dm * 2);
  _Float16* xn_h    = (_Float16*)alloc((size_t)Mrows * Dm * 2);
  _Float16* qkv_h   = (_Float16*)alloc((size_t)Mrows * 3 * Dm * 2);
  _Float16* aout_h  = (_Float16*)alloc((size_t)Mrows * Dm * 2);
  float*    x1      = (float*)   alloc((size_t)Mrows * Dm * 4);
  _Float16* h_h     = (_Float16*)alloc((size_t)Mrows * Dff * 2);

  // 1) weights fp32 -> fp16
  cvt_f32_f16_kernel<<<4096, 256, 0, stream>>>(w_qkv, w_qkv_h, Dm * 3 * Dm);
  cvt_f32_f16_kernel<<<4096, 256, 0, stream>>>(w_out, w_out_h, Dm * Dm);
  cvt_f32_f16_kernel<<<4096, 256, 0, stream>>>(w_ff1, w_ff1_h, Dm * Dff);
  cvt_f32_f16_kernel<<<4096, 256, 0, stream>>>(w_ff2, w_ff2_h, Dff * Dm);

  // 2) LN1 -> f16
  layernorm_f16_kernel<<<Mrows, 256, 0, stream>>>(x, ln1_g, ln1_b, xn_h, Dm);

  // 3) qkv = xn @ w_qkv + b_qkv        [4096 x 6144]
  gemm_f16_kernel<<<(Mrows / 64) * (3 * Dm / 128), 128, 0, stream>>>(
      xn_h, w_qkv_h, b_qkv, nullptr, nullptr, qkv_h, Mrows, 3 * Dm, Dm, 0);

  // 4) RoPE on q,k halves (in place)
  rope_kernel<<<(Mrows * Nh * (Dh / 2) + 255) / 256, 256, 0, stream>>>(
      qkv_h, cosT, sinT);

  // 5) causal flash attention -> aout_h
  flash_attn_kernel<<<dim3(Seq / 128, Nh, Bsz), 256, 0, stream>>>(qkv_h, aout_h);

  // 6) x1 = x + aout @ w_out + b_out   [4096 x 2048]
  gemm_f16_kernel<<<(Mrows / 64) * (Dm / 128), 128, 0, stream>>>(
      aout_h, w_out_h, b_out, x, x1, nullptr, Mrows, Dm, Dm, 0);

  // 7) LN2 -> f16
  layernorm_f16_kernel<<<Mrows, 256, 0, stream>>>(x1, ln2_g, ln2_b, xn_h, Dm);

  // 8) h = gelu(xn2 @ w_ff1 + b_ff1)   [4096 x 8192] -> f16
  gemm_f16_kernel<<<(Mrows / 64) * (Dff / 128), 128, 0, stream>>>(
      xn_h, w_ff1_h, b_ff1, nullptr, nullptr, h_h, Mrows, Dff, Dm, 1);

  // 9) out = x1 + h @ w_ff2 + b_ff2    [4096 x 2048] -> fp32 d_out
  gemm_f16_kernel<<<(Mrows / 64) * (Dm / 128), 128, 0, stream>>>(
      h_h, w_ff2_h, b_ff2, x1, (float*)d_out, nullptr, Mrows, Dm, Dff, 0);
}
